// TaskAlignedAssigner_53369263620710
// MI455X (gfx1250) — matrix operand order repaired and checked by reading, not compiled, not run on hardware
//
#include <hip/hip_runtime.h>
#include <hip/hip_bf16.h>
#include <stdint.h>

// Problem constants (from reference)
#define BB   16
#define NN   100
#define MM   8400      // 80*80 + 40*40 + 20*20
#define CC   80
#define TOPKK 13
#define EPSF 1e-9f

typedef __attribute__((ext_vector_type(16))) _Float16 v16h;
typedef __attribute__((ext_vector_type(8)))  float    v8f;

__device__ __forceinline__ float iou_box(float g0,float g1,float g2,float g3,
                                         float p0,float p1,float p2,float p3) {
  float lx = fmaxf(g0,p0), ly = fmaxf(g1,p1);
  float rx = fminf(g2,p2), ry = fminf(g3,p3);
  float w = fmaxf(rx-lx,0.f), h = fmaxf(ry-ly,0.f);
  float inter = w*h;
  float ag = (g2-g0)*(g3-g1);
  float ap = (p2-p0)*(p3-p1);
  return fmaxf(inter / (ag+ap-inter+EPSF), 0.f);
}

// -------------------------------------------------------------------------
// K1: cls[b,n,m] = pred_scores[b,m,lab[b,n]] computed as OneHot(N x 80) @
//     Scores^T(80 x M) with V_WMMA_F32_16X16X32_F16. One wave per (b, 16-col
//     m-tile); B fragments for all 3 K-chunks stay resident; 7 n-tiles reuse
//     them (21 wmma / wave).
// -------------------------------------------------------------------------
__global__ __launch_bounds__(32)
void k_cls_wmma(const float* __restrict__ scores, const int* __restrict__ lab,
                float* __restrict__ cls) {
  const int MT   = MM / 16;              // 525 column tiles
  const int b    = blockIdx.x / MT;
  const int mt   = blockIdx.x % MT;
  const int lane = threadIdx.x;          // 0..31 (wave32)
  const int col  = lane & 15;
  const int kh   = lane >> 4;            // K-half select
  const int m    = mt * 16 + col;

  // B fragment: element e of v16h holds B[k = ch*32 + kh*16 + e, col]
  // = scores[b, m, k]  -> 16 contiguous floats per lane per chunk.
  const float* srow = scores + ((size_t)b * MM + m) * CC;
  v16h bf[3];
#pragma unroll
  for (int ch = 0; ch < 3; ++ch) {
    v16h t;
#pragma unroll
    for (int e = 0; e < 16; ++e) {
      int k = ch * 32 + kh * 16 + e;
      float v = (k < CC) ? srow[k] : 0.f;
      t[e] = (_Float16)v;
    }
    bf[ch] = t;
  }

  const int row = lane & 15;             // A-matrix M index for this lane
  for (int nt = 0; nt < 7; ++nt) {       // 112 >= 100 rows
    int n_row  = nt * 16 + row;
    int labrow = (n_row < NN) ? lab[b * NN + n_row] : -1;

    v8f acc = {};
#pragma unroll
    for (int ch = 0; ch < 3; ++ch) {
      // A fragment (this code's contiguous k order): element e holds
      // K = kh*16 + e  (+ ch*32 globally), matching B
      v16h a;
#pragma unroll
      for (int e = 0; e < 16; ++e) {
        int K = e + kh * 16 + ch * 32;
        a[e] = (labrow == K) ? (_Float16)1.0f : (_Float16)0.0f;
      }
      acc = __builtin_amdgcn_wmma_f32_16x16x32_f16(
          false, a, false, bf[ch], (short)0, acc, false, false);
    }
    // D layout: VGPR r, lanes<16 -> M=r, lanes>=16 -> M=r+8 ; N = lane&15
#pragma unroll
    for (int r = 0; r < 8; ++r) {
      int n = nt * 16 + r + kh * 8;
      if (n < NN) cls[((size_t)b * NN + n) * MM + m] = acc[r];
    }
  }
}

// -------------------------------------------------------------------------
// K2: per (b,n): top-13 of align*in_gts via 13 block-argmax rounds (stable,
//     first-index tie-break == jax.lax.top_k), scatter bit n into per-(b,m)
//     128-bit mask iff in_gts && mask_gt.
// -------------------------------------------------------------------------
__global__ __launch_bounds__(256)
void k_topk(const float* __restrict__ cls, const float* __restrict__ pb,
            const float* __restrict__ ap,  const float* __restrict__ gt,
            const float* __restrict__ mgt, unsigned* __restrict__ colmask) {
  const int b = blockIdx.x / NN;
  const int n = blockIdx.x % NN;
  const int tid = threadIdx.x;
  constexpr int ITER = (MM + 255) / 256;   // 33

  __shared__ float sv[256];
  __shared__ int   si[256];
  __shared__ int   sbest;
  __shared__ int   sel[TOPKK];

  const float g0 = gt[(b*NN+n)*4+0], g1 = gt[(b*NN+n)*4+1];
  const float g2 = gt[(b*NN+n)*4+2], g3 = gt[(b*NN+n)*4+3];

  const float* crow = cls + ((size_t)b * NN + n) * MM;

  float val[ITER];
#pragma unroll 4
  for (int r = 0; r < ITER; ++r) {
    int m = tid + r * 256;
    float v = -1.f;
    if (m < MM) {
      float p0 = pb[((size_t)b*MM+m)*4+0], p1 = pb[((size_t)b*MM+m)*4+1];
      float p2 = pb[((size_t)b*MM+m)*4+2], p3 = pb[((size_t)b*MM+m)*4+3];
      float iou = iou_box(g0,g1,g2,g3,p0,p1,p2,p3);
      float i2 = iou*iou;
      float align = crow[m] * (i2*i2*i2);
      float apx = ap[2*m], apy = ap[2*m+1];
      float dmin = fminf(fminf(apx-g0, apy-g1), fminf(g2-apx, g3-apy));
      v = (dmin > EPSF) ? align : 0.f;
    }
    val[r] = v;
  }

  for (int t = 0; t < TOPKK; ++t) {
    float bv = -2.f; int bi = 0x7fffffff;
#pragma unroll 4
    for (int r = 0; r < ITER; ++r) {
      if (val[r] > bv) { bv = val[r]; bi = tid + r * 256; }
    }
    sv[tid] = bv; si[tid] = bi;
    __syncthreads();
    for (int s = 128; s > 0; s >>= 1) {
      if (tid < s) {
        if (sv[tid+s] > sv[tid] ||
            (sv[tid+s] == sv[tid] && si[tid+s] < si[tid])) {
          sv[tid] = sv[tid+s]; si[tid] = si[tid+s];
        }
      }
      __syncthreads();
    }
    if (tid == 0) { sbest = si[0]; sel[t] = si[0]; }
    __syncthreads();
    int bm = sbest;
    if ((bm & 255) == tid) val[bm >> 8] = -2.f;   // remove from candidates
    __syncthreads();
  }

  if (tid < TOPKK) {
    if (mgt[b*NN+n] != 0.f) {
      int m = sel[tid];
      float apx = ap[2*m], apy = ap[2*m+1];
      float dmin = fminf(fminf(apx-g0, apy-g1), fminf(g2-apx, g3-apy));
      if (dmin > EPSF)
        atomicOr(&colmask[((size_t)b*MM+m)*4 + (n>>5)], 1u << (n & 31));
    }
  }
}

// -------------------------------------------------------------------------
// K3: per (b,m): resolve multi-assignment (argmax IoU over all N), record
//     winning GT, its align value, and per-GT pa/po maxes (uint-bit atomicMax,
//     valid for non-negative floats).
// -------------------------------------------------------------------------
__global__ __launch_bounds__(256)
void k_fix(const float* __restrict__ cls, const float* __restrict__ pb,
           const float* __restrict__ gt, const unsigned* __restrict__ colmask,
           int* __restrict__ tgt, float* __restrict__ alignsel,
           unsigned* __restrict__ pa, unsigned* __restrict__ po) {
  int i = blockIdx.x * 256 + threadIdx.x;
  if (i >= BB * MM) return;
  int b = i / MM, m = i % MM;

  unsigned w0 = colmask[(size_t)i*4+0], w1 = colmask[(size_t)i*4+1];
  unsigned w2 = colmask[(size_t)i*4+2], w3 = colmask[(size_t)i*4+3];
  int fg0 = __popc(w0) + __popc(w1) + __popc(w2) + __popc(w3);

  float p0 = pb[(size_t)i*4+0], p1 = pb[(size_t)i*4+1];
  float p2 = pb[(size_t)i*4+2], p3 = pb[(size_t)i*4+3];

  int n0 = -1;
  if (fg0 == 1) {
    n0 = w0 ? (__ffs(w0)-1) : w1 ? (32+__ffs(w1)-1)
       : w2 ? (64+__ffs(w2)-1) : (96+__ffs(w3)-1);
  } else if (fg0 > 1) {
    float best = -1.f;
    for (int n = 0; n < NN; ++n) {
      const float* g = gt + ((size_t)b*NN + n) * 4;
      float iou = iou_box(g[0],g[1],g[2],g[3],p0,p1,p2,p3);
      if (iou > best) { best = iou; n0 = n; }   // first-tie == jnp.argmax
    }
  }
  tgt[i] = n0;

  float al = 0.f;
  if (n0 >= 0) {
    const float* g = gt + ((size_t)b*NN + n0) * 4;
    float iou = iou_box(g[0],g[1],g[2],g[3],p0,p1,p2,p3);
    float i2 = iou*iou;
    al = cls[((size_t)b*NN + n0)*MM + m] * (i2*i2*i2);
    atomicMax(&pa[b*128 + n0], __float_as_uint(al));
    atomicMax(&po[b*128 + n0], __float_as_uint(iou));
  }
  alignsel[i] = al;
}

// -------------------------------------------------------------------------
// K4: per (b,m): tl, tb, fg, norm.  (Reference gathers tb at index 0 for
//     background anchors.)
// -------------------------------------------------------------------------
__global__ __launch_bounds__(256)
void k_head(const int* __restrict__ tgt, const float* __restrict__ alignsel,
            const unsigned* __restrict__ pa, const unsigned* __restrict__ po,
            const int* __restrict__ lab, const float* __restrict__ gt,
            float* __restrict__ out, int* __restrict__ tlbuf,
            float* __restrict__ normbuf) {
  int i = blockIdx.x * 256 + threadIdx.x;
  if (i >= BB * MM) return;
  int b = i / MM;
  const size_t BM = (size_t)BB * MM;

  int t  = tgt[i];
  int fg = (t >= 0);
  int ng = fg ? t : 0;
  int tl = fg ? lab[b*NN + ng] : CC;

  float norm = 0.f;
  if (fg) {
    float paf = __uint_as_float(pa[b*128 + t]);
    float pof = __uint_as_float(po[b*128 + t]);
    norm = alignsel[i] * pof / (paf + EPSF);
  }

  out[i] = (float)tl;                                  // tl
  const float* g = gt + ((size_t)b*NN + ng) * 4;       // tb
  float* tb = out + BM + (size_t)i * 4;
  tb[0] = g[0]; tb[1] = g[1]; tb[2] = g[2]; tb[3] = g[3];
  out[BM*85 + i] = fg ? 1.f : 0.f;                     // fg>0

  tlbuf[i] = tl; normbuf[i] = norm;
}

// K5: ts = one_hot(tl, 81)[..., :80] * norm   (coalesced, 1 thread / element)
__global__ __launch_bounds__(256)
void k_ts(const int* __restrict__ tlbuf, const float* __restrict__ normbuf,
          float* __restrict__ out) {
  int i = blockIdx.x * 256 + threadIdx.x;
  if (i >= BB * MM * CC) return;
  int bm = i / CC, c = i % CC;
  const size_t BM = (size_t)BB * MM;
  out[BM*5 + i] = (c == tlbuf[bm]) ? normbuf[bm] : 0.f;
}

extern "C" void kernel_launch(void* const* d_in, const int* in_sizes, int n_in,
                              void* d_out, int out_size, void* d_ws, size_t ws_size,
                              hipStream_t stream) {
  const float* pred_scores   = (const float*)d_in[0];
  const float* pred_bboxes   = (const float*)d_in[1];
  const float* anchor_points = (const float*)d_in[2];
  const float* gt_bboxes     = (const float*)d_in[3];
  const int*   gt_labels     = (const int*)  d_in[4];
  const float* mask_gt       = (const float*)d_in[5];
  float* out = (float*)d_out;

  // Workspace carve-out (256B aligned)
  char* w = (char*)d_ws;
  auto carve = [&](size_t bytes) { char* p = w; w += (bytes + 255) & ~(size_t)255; return p; };
  float*    cls      = (float*)   carve((size_t)BB*NN*MM*sizeof(float));     // ~53.8 MB
  unsigned* colmask  = (unsigned*)carve((size_t)BB*MM*4*sizeof(unsigned));   // 2.15 MB
  int*      tgt      = (int*)     carve((size_t)BB*MM*sizeof(int));
  float*    alignsel = (float*)   carve((size_t)BB*MM*sizeof(float));
  int*      tlbuf    = (int*)     carve((size_t)BB*MM*sizeof(int));
  float*    normbuf  = (float*)   carve((size_t)BB*MM*sizeof(float));
  unsigned* pa       = (unsigned*)carve((size_t)BB*128*sizeof(unsigned));
  unsigned* po       = (unsigned*)carve((size_t)BB*128*sizeof(unsigned));

  // Atomically-accumulated buffers must start at zero every call
  hipMemsetAsync(colmask, 0, (size_t)BB*MM*4*sizeof(unsigned), stream);
  hipMemsetAsync(pa,      0, (size_t)BB*128*sizeof(unsigned),  stream);
  hipMemsetAsync(po,      0, (size_t)BB*128*sizeof(unsigned),  stream);

  k_cls_wmma<<<BB * (MM/16), 32, 0, stream>>>(pred_scores, gt_labels, cls);
  k_topk<<<BB * NN, 256, 0, stream>>>(cls, pred_bboxes, anchor_points,
                                      gt_bboxes, mask_gt, colmask);
  k_fix<<<(BB*MM + 255)/256, 256, 0, stream>>>(cls, pred_bboxes, gt_bboxes,
                                               colmask, tgt, alignsel, pa, po);
  k_head<<<(BB*MM + 255)/256, 256, 0, stream>>>(tgt, alignsel, pa, po,
                                                gt_labels, gt_bboxes, out,
                                                tlbuf, normbuf);
  k_ts<<<(BB*MM*CC + 255)/256, 256, 0, stream>>>(tlbuf, normbuf, out);
}
